// Matchmaker_context_v2_42674795053890
// MI455X (gfx1250) — matrix operand order repaired and checked by reading, not compiled, not run on hardware
//
#include <hip/hip_runtime.h>

typedef __attribute__((ext_vector_type(16))) _Float16 v16h;
typedef __attribute__((ext_vector_type(8)))  _Float16 v8h;
typedef __attribute__((ext_vector_type(4)))  _Float16 v4h;
typedef __attribute__((ext_vector_type(8)))  float    v8f;

// ------------------------- constants (KNRM kernels) -------------------------
__constant__ float c_MU[11] = {1.0f, 0.9f, 0.7f, 0.5f, 0.3f, 0.1f,
                               -0.1f, -0.3f, -0.5f, -0.7f, -0.9f};
// 1/(2*sigma^2): sigma0 = 1e-4 -> 5e7 ; others sigma = 0.1 -> 50
__constant__ float c_I2S2[11] = {5.0e7f, 50.f, 50.f, 50.f, 50.f, 50.f,
                                 50.f, 50.f, 50.f, 50.f, 50.f};

// ------------------------- generic WMMA GEMM -------------------------------
// C[M,N] = A[M,K] * op(B) (+bias)(+relu), fp32 in/out, f16 WMMA compute.
// Block: 256 threads = 8 waves, tile BMT x 128, K-step 32.
// Wave (wm = w>>2, wn = w&3) computes a (16*AF) x 32 sub-tile (AF = BMT/32).
// B tile is stored transposed in LDS (Bt[n][k]) so fragments are contiguous
// 16B chunks -> ds_load_b128. Fast path uses float4 global loads; edge path is
// branchless (clamped address + select). CONV mode: A row m valid for
// gk < (convT - m%convT)*convE with row stride lda (implicit causal im2col).
// Columns in [N, ldc) are written as zeros (used to zero-pad ff1/conv bufs).
#define BK 32
#define BN 128

template <int BMT, bool TRANSB, bool BIAS, bool RELU, bool CONV>
__global__ __launch_bounds__(256) void gemm_wmma_kernel(
    const float* __restrict__ A, const float* __restrict__ B,
    const float* __restrict__ bias, float* __restrict__ C,
    int M, int N, int Kdim, int Kb, int lda, int ldb, int ldc,
    long long aBatch, long long bBatch, long long cBatch,
    int convT, int convE)
{
  constexpr int AF = BMT / 32;  // A fragments per wave (1 or 2)
  __shared__ alignas(16) _Float16 As[BMT][BK + 8];
  __shared__ alignas(16) _Float16 Bt[BN][BK + 8];  // [n][k]

  const int tid  = threadIdx.x;
  const int wave = tid >> 5, lane = tid & 31;
  const int half = lane >> 4, l16 = lane & 15;
  const int wm = wave >> 2, wn = wave & 3;

  const int m0 = blockIdx.y * BMT;
  const int n0 = blockIdx.x * BN;
  const float* Ab = A + (long long)blockIdx.z * aBatch;
  const float* Bb = B + (long long)blockIdx.z * bBatch;
  float*       Cb = C + (long long)blockIdx.z * cBatch;

  const bool lda4 = ((lda & 3) == 0);
  const bool ldb4 = ((ldb & 3) == 0);

  v8f acc[AF][2];
  const v8f vz = {0.f, 0.f, 0.f, 0.f, 0.f, 0.f, 0.f, 0.f};
#pragma unroll
  for (int a = 0; a < AF; ++a) { acc[a][0] = vz; acc[a][1] = vz; }

  const int nK = (Kdim + BK - 1) / BK;
  for (int kt = 0; kt < nK; ++kt) {
    const int k0 = kt * BK;

    // ---------------- stage A tile (BMT x 32) ----------------
    bool fullA = (m0 + BMT <= M) && (k0 + BK <= Kdim) && lda4;
    if (CONV) {
      int t0 = m0 % convT;
      int lim = (t0 + BMT <= convT) ? (convT - (t0 + BMT - 1)) * convE : convE;
      fullA = fullA && (k0 + BK <= lim);
    }
    if (fullA) {
#pragma unroll
      for (int i = 0; i < AF; ++i) {
        int r = (tid >> 3) + i * 32;
        int c = (tid & 7) * 4;
        float4 f = *(const float4*)(Ab + (long long)(m0 + r) * lda + (k0 + c));
        v4h h = {(_Float16)f.x, (_Float16)f.y, (_Float16)f.z, (_Float16)f.w};
        *(v4h*)&As[r][c] = h;
      }
    } else {
#pragma unroll
      for (int i = 0; i < AF * 4; ++i) {
        int idx = tid + i * 256;
        int r = idx >> 5, c = idx & 31;
        int m = m0 + r, gk = k0 + c;
        int klim = Kdim;
        if (CONV) {
          int t = (m < M) ? (m % convT) : 0;
          int l = (convT - t) * convE;
          klim = l < Kdim ? l : Kdim;
        }
        bool ok = (m < M) && (gk < klim);
        long long off = ok ? ((long long)m * lda + gk) : 0;
        float v = Ab[off];
        As[r][c] = (_Float16)(ok ? v : 0.f);
      }
    }

    // ---------------- stage B tile (32 x 128) -> Bt[n][k] ----------------
    const bool fullB = (n0 + BN <= N) && (k0 + BK <= Kb) && ldb4;
    if (fullB) {
      if (TRANSB) {
#pragma unroll
        for (int i = 0; i < 4; ++i) {
          int n  = (tid >> 3) + i * 32;
          int k4 = (tid & 7) * 4;
          float4 f = *(const float4*)(Bb + (long long)(n0 + n) * ldb + (k0 + k4));
          v4h h = {(_Float16)f.x, (_Float16)f.y, (_Float16)f.z, (_Float16)f.w};
          *(v4h*)&Bt[n][k4] = h;
        }
      } else {
#pragma unroll
        for (int i = 0; i < 4; ++i) {
          int fid = tid + i * 256;        // 1024 float4's
          int k  = fid >> 5;
          int c4 = (fid & 31) * 4;
          float4 f = *(const float4*)(Bb + (long long)(k0 + k) * ldb + (n0 + c4));
          Bt[c4 + 0][k] = (_Float16)f.x;
          Bt[c4 + 1][k] = (_Float16)f.y;
          Bt[c4 + 2][k] = (_Float16)f.z;
          Bt[c4 + 3][k] = (_Float16)f.w;
        }
      }
    } else {
#pragma unroll
      for (int i = 0; i < 16; ++i) {
        int idx = tid + i * 256;
        int n = idx & 127, k = idx >> 7;
        int gn = n0 + n, gk = k0 + k;
        bool ok = (gn < N) && (gk < Kb);
        long long off =
            ok ? (TRANSB ? (long long)gn * ldb + gk : (long long)gk * ldb + gn) : 0;
        float v = Bb[off];
        Bt[n][k] = (_Float16)(ok ? v : 0.f);
      }
    }
    __syncthreads();

    // ---------------- fragments + WMMA ----------------
    v16h af[AF], bf[2];
#pragma unroll
    for (int a = 0; a < AF; ++a) {
      int ar = wm * (AF * 16) + a * 16 + l16;
      v8h x0 = *(const v8h*)&As[ar][half * 8];
      v8h x1 = *(const v8h*)&As[ar][16 + half * 8];
      af[a] = __builtin_shufflevector(x0, x1, 0, 1, 2, 3, 4, 5, 6, 7,
                                      8, 9, 10, 11, 12, 13, 14, 15);
    }
#pragma unroll
    for (int b = 0; b < 2; ++b) {
      int bc = wn * 32 + b * 16 + l16;
      v8h x0 = *(const v8h*)&Bt[bc][half * 8];
      v8h x1 = *(const v8h*)&Bt[bc][16 + half * 8];
      bf[b] = __builtin_shufflevector(x0, x1, 0, 1, 2, 3, 4, 5, 6, 7,
                                      8, 9, 10, 11, 12, 13, 14, 15);
    }
#pragma unroll
    for (int a = 0; a < AF; ++a)
#pragma unroll
      for (int b = 0; b < 2; ++b)
        acc[a][b] = __builtin_amdgcn_wmma_f32_16x16x32_f16(
            false, af[a], false, bf[b], (short)0, acc[a][b], false, false);
    __syncthreads();
  }

  // ---------------- store (cols in [N, ldc) written as zero) ----------------
#pragma unroll
  for (int b = 0; b < 2; ++b) {
    int ccol = n0 + wn * 32 + b * 16 + l16;
    if (ccol < ldc) {
      float bv = 0.f;
      if (BIAS && ccol < N) bv = bias[ccol];
#pragma unroll
      for (int a = 0; a < AF; ++a) {
#pragma unroll
        for (int g = 0; g < 8; ++g) {
          int crow = m0 + wm * (AF * 16) + a * 16 + g + 8 * half;
          if (crow < M) {
            float v = (ccol < N) ? (acc[a][b][g] + bv) : 0.f;
            if (RELU) v = v > 0.f ? v : 0.f;
            Cb[(long long)crow * ldc + ccol] = v;
          }
        }
      }
    }
  }
}

// ------------------------- elementwise / small kernels ---------------------
__global__ void add_pos_kernel(const float* __restrict__ x,
                               float* __restrict__ out, int T, long long total)
{
  long long idx = (long long)blockIdx.x * 256 + threadIdx.x;
  if (idx >= total) return;
  int e = (int)(idx & 511);
  int t = (int)((idx >> 9) % T);
  const float c = -9.21034037197618f / 255.f;  // -ln(10000)/(nts-1)
  float pos = (e < 256) ? __sinf((float)t * __expf(c * (float)e))
                        : __cosf((float)t * __expf(c * (float)(e - 256)));
  out[idx] = x[idx] + pos;
}

// out = LN(a + b) over E=512 (in-place safe on a)
__global__ void ln_res_kernel(const float* a, const float* b,
                              const float* __restrict__ g,
                              const float* __restrict__ bb, float* out)
{
  __shared__ float red[256];
  long long base = (long long)blockIdx.x * 512;
  int tid = threadIdx.x;
  float z0 = a[base + tid] + b[base + tid];
  float z1 = a[base + 256 + tid] + b[base + 256 + tid];
  red[tid] = z0 + z1;
  __syncthreads();
  for (int s = 128; s > 0; s >>= 1) { if (tid < s) red[tid] += red[tid + s]; __syncthreads(); }
  float mean = red[0] * (1.f / 512.f);
  __syncthreads();
  red[tid] = (z0 - mean) * (z0 - mean) + (z1 - mean) * (z1 - mean);
  __syncthreads();
  for (int s = 128; s > 0; s >>= 1) { if (tid < s) red[tid] += red[tid + s]; __syncthreads(); }
  float inv = 1.f / (sqrtf(red[0] * (1.f / 512.f)) + 1e-6f);
  out[base + tid]       = g[tid] * (z0 - mean) * inv + bb[tid];
  out[base + 256 + tid] = g[256 + tid] * (z1 - mean) * inv + bb[256 + tid];
}

// per (b,t,h): rank-1 masked softmax attention (head dim = 1)
__global__ void attention_kernel(const float* __restrict__ comb,
                                 const float* __restrict__ mask,
                                 float* __restrict__ att, int T)
{
  int tid = threadIdx.x;
  int h = tid & 31;
  long long row = (long long)blockIdx.x * 4 + (tid >> 5);  // b*T + i
  int b = (int)(row / T);
  float q = comb[row * 96 + h] * 0.25f;  // 1/sqrt(E/N_HEADS)
  const float* kb = comb + (long long)b * T * 96;
  const float* mb = mask + (long long)b * T;
  float S = 0.f, Spm = 0.f, Nv = 0.f;
#pragma unroll 4
  for (int j = 0; j < T; ++j) {
    float m  = mb[j];
    float kk = kb[(long long)j * 96 + 32 + h];
    float vv = kb[(long long)j * 96 + 64 + h];
    float p  = __expf(q * kk * m);
    S += p;
    float pm = p * m;
    Spm += pm;
    Nv  += pm * vv;
  }
  att[row * 32 + h] = Nv / (Spm + S * 1e-13f);
}

// enc = LN(atto + ff); cn = normalize(concat[x, enc]) over 1024 dims
__global__ void enc_cat_norm_kernel(const float* __restrict__ atto,
                                    const float* __restrict__ ff,
                                    const float* __restrict__ x,
                                    const float* __restrict__ g,
                                    const float* __restrict__ bb,
                                    float* __restrict__ cn)
{
  __shared__ float red[256];
  long long base = (long long)blockIdx.x * 512;
  int tid = threadIdx.x;
  float z0 = atto[base + tid] + ff[base + tid];
  float z1 = atto[base + 256 + tid] + ff[base + 256 + tid];
  red[tid] = z0 + z1;
  __syncthreads();
  for (int s = 128; s > 0; s >>= 1) { if (tid < s) red[tid] += red[tid + s]; __syncthreads(); }
  float mean = red[0] * (1.f / 512.f);
  __syncthreads();
  red[tid] = (z0 - mean) * (z0 - mean) + (z1 - mean) * (z1 - mean);
  __syncthreads();
  for (int s = 128; s > 0; s >>= 1) { if (tid < s) red[tid] += red[tid + s]; __syncthreads(); }
  float inv = 1.f / (sqrtf(red[0] * (1.f / 512.f)) + 1e-6f);
  __syncthreads();
  float e0 = g[tid] * (z0 - mean) * inv + bb[tid];
  float e1 = g[256 + tid] * (z1 - mean) * inv + bb[256 + tid];
  float x0 = x[base + tid], x1 = x[base + 256 + tid];
  red[tid] = e0 * e0 + e1 * e1 + x0 * x0 + x1 * x1;
  __syncthreads();
  for (int s = 128; s > 0; s >>= 1) { if (tid < s) red[tid] += red[tid + s]; __syncthreads(); }
  float nrm = 1.f / (sqrtf(red[0]) + 1e-13f);
  long long ob = (long long)blockIdx.x * 1024;
  cn[ob + tid]       = x0 * nrm;
  cn[ob + 256 + tid] = x1 * nrm;
  cn[ob + 512 + tid] = e0 * nrm;
  cn[ob + 768 + tid] = e1 * nrm;
}

// in-place row normalization over 320-stride conv rows (cols 300.. are zero)
__global__ void conv_norm_kernel(float* buf)
{
  __shared__ float red[256];
  long long base = (long long)blockIdx.x * 320;
  int tid = threadIdx.x;
  float v0 = buf[base + tid];
  float v1 = (tid < 64) ? buf[base + 256 + tid] : 0.f;
  red[tid] = v0 * v0 + v1 * v1;
  __syncthreads();
  for (int s = 128; s > 0; s >>= 1) { if (tid < s) red[tid] += red[tid + s]; __syncthreads(); }
  float nrm = 1.f / (sqrtf(red[0]) + 1e-13f);
  buf[base + tid] = v0 * nrm;
  if (tid < 64) buf[base + 256 + tid] = v1 * nrm;
}

// pack conv weights (O=300, E=512, kw) -> row-major (kw*512, 300)
__global__ void pack_w_kernel(const float* __restrict__ w,
                              float* __restrict__ wp, int kw)
{
  int idx = blockIdx.x * 256 + threadIdx.x;
  int total = kw * 512 * 300;
  if (idx >= total) return;
  int o = idx % 300;
  int p = idx / 300;
  int e = p % 512;
  int i = p / 512;
  wp[idx] = w[((long long)o * 512 + e) * kw + i];
}

// kq[b,q,k] = qm[b,q] * sum_d exp(-(m-mu_k)^2/(2 s_k^2)) * dm[b,d]
__global__ void pool_kq_kernel(const float* __restrict__ match,
                               const float* __restrict__ qm,
                               const float* __restrict__ dm,
                               float* __restrict__ kq)
{
  __shared__ float sh[11][257];
  int b = blockIdx.x >> 5;  // Q = 32
  int q = blockIdx.x & 31;
  int tid = threadIdx.x;
  float acc[11] = {};
  const float* mr  = match + ((long long)b * 32 + q) * 512;
  const float* dmb = dm + (long long)b * 512;
  for (int d = tid; d < 512; d += 256) {
    float m = mr[d], w = dmb[d];
    if (w != 0.f) {
#pragma unroll
      for (int k = 0; k < 11; ++k) {
        float df = m - c_MU[k];
        acc[k] += __expf(-df * df * c_I2S2[k]) * w;
      }
    }
  }
#pragma unroll
  for (int k = 0; k < 11; ++k) sh[k][tid] = acc[k];
  __syncthreads();
  for (int s = 128; s > 0; s >>= 1) {
    if (tid < s)
#pragma unroll
      for (int k = 0; k < 11; ++k) sh[k][tid] += sh[k][tid + s];
    __syncthreads();
  }
  if (tid < 11) kq[((long long)b * 32 + q) * 11 + tid] = sh[tid][0] * qm[b * 32 + q];
}

// s[b,k] = sum_q log(max(kq,1e-10))*qm ; mn likewise with kq/doc_len
__global__ void pool_fin_kernel(const float* __restrict__ kq,
                                const float* __restrict__ qm,
                                const float* __restrict__ dm,
                                float* __restrict__ s_out,
                                float* __restrict__ m_out)
{
  __shared__ float red[256];
  int b = blockIdx.x, tid = threadIdx.x;
  red[tid] = dm[(long long)b * 512 + tid] + dm[(long long)b * 512 + 256 + tid];
  __syncthreads();
  for (int s = 128; s > 0; s >>= 1) { if (tid < s) red[tid] += red[tid + s]; __syncthreads(); }
  float dl = red[0];
  if (tid < 11) {
    float s = 0.f, mn = 0.f;
    for (int q = 0; q < 32; ++q) {
      float w = qm[b * 32 + q];
      float v = kq[((long long)b * 32 + q) * 11 + tid];
      s  += logf(fmaxf(v, 1e-10f)) * w;
      mn += logf(fmaxf(v / dl, 1e-10f)) * w;
    }
    s_out[b * 11 + tid] = s;
    m_out[b * 11 + tid] = mn;
  }
}

__global__ void mpd_kernel(const float* __restrict__ oto, float* __restrict__ mpd)
{
  int idx = blockIdx.x * 256 + threadIdx.x;  // b*512 + d
  if (idx >= 128 * 512) return;
  int b = idx >> 9, d = idx & 511;
  const float* p = oto + (long long)b * 32 * 512 + d;
  float mx = p[0];
  for (int q = 1; q < 32; ++q) mx = fmaxf(mx, p[(long long)q * 512]);
  mpd[idx] = mx;
}

__global__ void prox_kernel(const float* __restrict__ mpd,
                            const float* __restrict__ dm,
                            const float* __restrict__ scaler,
                            const float* __restrict__ pw,
                            const float* __restrict__ pb,
                            const float* __restrict__ pcw,
                            const float* __restrict__ pcb,
                            float* __restrict__ pout)
{
  __shared__ float c[513];
  __shared__ float pt[12][512];
  __shared__ float ps[12][11];
  __shared__ float plin[12];
  int b = blockIdx.x, tid = threadIdx.x;
  if (tid == 0) {
    c[0] = 0.f;
    float a = 0.f;
    for (int d = 0; d < 512; ++d) { a += mpd[(long long)b * 512 + d]; c[d + 1] = a; }
  }
  __syncthreads();
  for (int idx = tid; idx < 12 * 512; idx += 256) {
    int i = idx >> 9, d = idx & 511;
    int r = i + 2;  // PROX_RANGE = 2..13
    pt[i][d] = (d < 513 - r) ? (c[d + r] - c[d]) / (float)r * scaler[i] : 0.f;
  }
  __syncthreads();
  if (tid < 132) {
    int i = tid / 11, k = tid % 11;
    float acc = 0.f;
    for (int d = 0; d < 512; ++d) {
      float df = pt[i][d] - c_MU[k];
      acc += __expf(-df * df * c_I2S2[k]) * dm[(long long)b * 512 + d];
    }
    ps[i][k] = acc;
  }
  __syncthreads();
  if (tid < 12) {
    float a = pb[tid];
    for (int k = 0; k < 11; ++k) a += ps[tid][k] * pw[tid * 11 + k];
    plin[tid] = a;
  }
  __syncthreads();
  if (tid == 0) {
    float a = pcb[0];
    for (int i = 0; i < 12; ++i) a += plin[i] * pcw[i];
    pout[b] = a;
  }
}

__global__ void final_kernel(const float* __restrict__ oto_s,
                             const float* __restrict__ oto_m,
                             const float* __restrict__ ngs,
                             const float* __restrict__ ngm,
                             const float* __restrict__ pout,
                             const float* __restrict__ los_w, const float* __restrict__ los_b,
                             const float* __restrict__ lom_w, const float* __restrict__ lom_b,
                             const float* __restrict__ ls_w,  const float* __restrict__ ls_b,
                             const float* __restrict__ lm_w,  const float* __restrict__ lm_b,
                             const float* __restrict__ comb_w, float* __restrict__ out)
{
  int b = threadIdx.x;  // 128 threads
  float f0 = los_b[0], f1 = lom_b[0], f2 = ls_b[0], f3 = lm_b[0];
  for (int k = 0; k < 11; ++k) {
    f0 += oto_s[b * 11 + k] * los_w[k];
    f1 += oto_m[b * 11 + k] * lom_w[k];
  }
  for (int w = 0; w < 3; ++w)
    for (int k = 0; k < 11; ++k) {
      f2 += ngs[(w * 128 + b) * 11 + k] * ls_w[w * 11 + k];
      f3 += ngm[(w * 128 + b) * 11 + k] * lm_w[w * 11 + k];
    }
  float f4 = pout[b];
  out[b] = f0 * comb_w[0] + f1 * comb_w[1] + f2 * comb_w[2] +
           f3 * comb_w[3] + f4 * comb_w[4];
}

// ------------------------- host orchestration ------------------------------
static void run_encoder(const float* x, const float* mask, long long M, int T,
                        const float* ff_w1, const float* ff_b1,
                        const float* ff_w2, const float* ff_b2,
                        const float* ffln_g, const float* ffln_b,
                        const float* att_cw, const float* att_cb,
                        const float* att_ow, const float* att_ob,
                        const float* ln_g, const float* ln_b,
                        float* bufA, float* bufC, float* bufB, float* bufAtt,
                        float* cn, hipStream_t stream)
{
  long long total = M * 512;
  add_pos_kernel<<<(unsigned)((total + 255) / 256), 256, 0, stream>>>(x, bufA, T, total);

  // ff1 = relu(xpos @ ff_w1 + b1), stored at stride 128 with zero-padded cols
  dim3 g1(1, (unsigned)((M + 63) / 64), 1);
  gemm_wmma_kernel<64, false, true, true, false><<<g1, 256, 0, stream>>>(
      bufA, ff_w1, ff_b1, bufB, (int)M, 100, 512, 512, 512, 100, 128, 0, 0, 0, 0, 0);

  // ff2 = ff1 @ ff_w2 + b2 (K padded 100->128; A pad cols are zero, Kb=100)
  dim3 g2(4, (unsigned)((M + 63) / 64), 1);
  gemm_wmma_kernel<64, false, true, false, false><<<g2, 256, 0, stream>>>(
      bufB, ff_w2, ff_b2, bufC, (int)M, 512, 128, 100, 128, 512, 512, 0, 0, 0, 0, 0);

  ln_res_kernel<<<(unsigned)M, 256, 0, stream>>>(bufC, bufA, ffln_g, ffln_b, bufC);

  // comb = ff @ att_cw + att_cb (N=96)
  dim3 g3(1, (unsigned)((M + 63) / 64), 1);
  gemm_wmma_kernel<64, false, true, false, false><<<g3, 256, 0, stream>>>(
      bufC, att_cw, att_cb, bufB, (int)M, 96, 512, 512, 512, 96, 96, 0, 0, 0, 0, 0);

  attention_kernel<<<(unsigned)(M / 4), 128, 0, stream>>>(bufB, mask, bufAtt, T);

  // atto = att @ att_ow + att_ob (K=32)
  dim3 g4(4, (unsigned)((M + 63) / 64), 1);
  gemm_wmma_kernel<64, false, true, false, false><<<g4, 256, 0, stream>>>(
      bufAtt, att_ow, att_ob, bufA, (int)M, 512, 32, 32, 32, 512, 512, 0, 0, 0, 0, 0);

  enc_cat_norm_kernel<<<(unsigned)M, 256, 0, stream>>>(bufA, bufC, x, ln_g, ln_b, cn);
}

extern "C" void kernel_launch(void* const* d_in, const int* in_sizes, int n_in,
                              void* d_out, int out_size, void* d_ws, size_t ws_size,
                              hipStream_t stream)
{
  (void)in_sizes; (void)n_in; (void)out_size; (void)ws_size;
  const float* qe  = (const float*)d_in[0];
  const float* de  = (const float*)d_in[1];
  const float* qm  = (const float*)d_in[2];
  const float* dmk = (const float*)d_in[3];
  const float* ff_w1 = (const float*)d_in[4];
  const float* ff_b1 = (const float*)d_in[5];
  const float* ff_w2 = (const float*)d_in[6];
  const float* ff_b2 = (const float*)d_in[7];
  const float* ffln_g = (const float*)d_in[8];
  const float* ffln_b = (const float*)d_in[9];
  const float* att_cw = (const float*)d_in[10];
  const float* att_cb = (const float*)d_in[11];
  const float* att_ow = (const float*)d_in[12];
  const float* att_ob = (const float*)d_in[13];
  const float* ln_g = (const float*)d_in[14];
  const float* ln_b = (const float*)d_in[15];
  const float* conv_w[3] = {(const float*)d_in[16], (const float*)d_in[18], (const float*)d_in[20]};
  const float* conv_b[3] = {(const float*)d_in[17], (const float*)d_in[19], (const float*)d_in[21]};
  const float* prox_scaler = (const float*)d_in[22];
  const float* prox_w = (const float*)d_in[23];
  const float* prox_b = (const float*)d_in[24];
  const float* prox_comb_w = (const float*)d_in[25];
  const float* prox_comb_b = (const float*)d_in[26];
  const float* ls_w = (const float*)d_in[27];
  const float* ls_b = (const float*)d_in[28];
  const float* lm_w = (const float*)d_in[29];
  const float* lm_b = (const float*)d_in[30];
  const float* los_w = (const float*)d_in[31];
  const float* los_b = (const float*)d_in[32];
  const float* lom_w = (const float*)d_in[33];
  const float* lom_b = (const float*)d_in[34];
  const float* comb_w = (const float*)d_in[35];
  float* out = (float*)d_out;

  const long long Mq = 128LL * 32, Md = 128LL * 512;

  // workspace bump allocator (fp32 units)
  size_t off = 0;
  float* Wf = (float*)d_ws;
  auto F = [&](size_t n) { float* p = Wf + off; off += n; return p; };
  float* bufA   = F(Md * 512);   // xpos / atto
  float* bufC   = F(Md * 512);   // ff2 / ff
  float* bufB   = F(Md * 128);   // ff1(pad128) / comb(96)
  float* bufAtt = F(Md * 32);
  float* qcn    = F(Mq * 1024);
  float* dcn    = F(Md * 1024);  // conv scratch aliases here after cosine GEMM
  float* oto    = F(Mq * 512);
  float* match  = F(Mq * 512);
  float* kq     = F(Mq * 11 + 4);
  float* oto_s  = F(128 * 11 + 4);
  float* oto_m  = F(128 * 11 + 4);
  float* ngs    = F(3 * 128 * 11 + 4);
  float* ngm    = F(3 * 128 * 11 + 4);
  float* mpd    = F(128 * 512);
  float* pout   = F(128);

  // conv scratch aliased inside dcn (free after the oto cosine GEMM)
  float* wp = dcn;                     // up to 2048*300
  float* cq = wp + 2048 * 300;         // Mq x 320 (zero-padded cols 300..319)
  float* cd = cq + Mq * 320;           // Md x 320

  // encoders -> normalized concat embeddings
  run_encoder(qe, qm, Mq, 32, ff_w1, ff_b1, ff_w2, ff_b2, ffln_g, ffln_b,
              att_cw, att_cb, att_ow, att_ob, ln_g, ln_b,
              bufA, bufC, bufB, bufAtt, qcn, stream);
  run_encoder(de, dmk, Md, 512, ff_w1, ff_b1, ff_w2, ff_b2, ffln_g, ffln_b,
              att_cw, att_cb, att_ow, att_ob, ln_g, ln_b,
              bufA, bufC, bufB, bufAtt, dcn, stream);

  // oto[b] = qcn[b] (32x1024) @ dcn[b]^T (512x1024)
  {
    dim3 gc(4, 1, 128);
    gemm_wmma_kernel<32, true, false, false, false><<<gc, 256, 0, stream>>>(
        qcn, dcn, nullptr, oto, 32, 512, 1024, 1024, 1024, 1024, 512,
        32LL * 1024, 512LL * 1024, 32LL * 512, 0, 0);
  }

  // KNRM pooling on oto
  pool_kq_kernel<<<128 * 32, 256, 0, stream>>>(oto, qm, dmk, kq);
  pool_fin_kernel<<<128, 256, 0, stream>>>(kq, qm, dmk, oto_s, oto_m);

  // proximity features
  mpd_kernel<<<(128 * 512) / 256, 256, 0, stream>>>(oto, mpd);
  prox_kernel<<<128, 256, 0, stream>>>(mpd, dmk, prox_scaler, prox_w, prox_b,
                                       prox_comb_w, prox_comb_b, pout);

  // conv n-gram paths (widths 2,3,4)
  for (int w = 0; w < 3; ++w) {
    int kw = w + 2;
    int Kc = kw * 512;
    pack_w_kernel<<<(kw * 512 * 300 + 255) / 256, 256, 0, stream>>>(conv_w[w], wp, kw);

    dim3 gq(3, (unsigned)((Mq + 63) / 64), 1);  // N=300, ldc=320 (zero pad)
    gemm_wmma_kernel<64, false, true, true, true><<<gq, 256, 0, stream>>>(
        qe, wp, conv_b[w], cq, (int)Mq, 300, Kc, Kc, 512, 300, 320, 0, 0, 0, 32, 512);
    dim3 gd(3, (unsigned)((Md + 63) / 64), 1);
    gemm_wmma_kernel<64, false, true, true, true><<<gd, 256, 0, stream>>>(
        de, wp, conv_b[w], cd, (int)Md, 300, Kc, Kc, 512, 300, 320, 0, 0, 0, 512, 512);

    conv_norm_kernel<<<(unsigned)Mq, 256, 0, stream>>>(cq);
    conv_norm_kernel<<<(unsigned)Md, 256, 0, stream>>>(cd);

    dim3 gm(4, 1, 128);  // match[b] = cq[b] (32x320) @ cd[b]^T (512x320)
    gemm_wmma_kernel<32, true, false, false, false><<<gm, 256, 0, stream>>>(
        cq, cd, nullptr, match, 32, 512, 320, 320, 320, 320, 512,
        32LL * 320, 512LL * 320, 32LL * 512, 0, 0);

    pool_kq_kernel<<<128 * 32, 256, 0, stream>>>(match, qm, dmk, kq);
    pool_fin_kernel<<<128, 256, 0, stream>>>(kq, qm, dmk,
                                             ngs + (size_t)w * 128 * 11,
                                             ngm + (size_t)w * 128 * 11);
  }

  // final feature combine -> (B,)
  final_kernel<<<1, 128, 0, stream>>>(oto_s, oto_m, ngs, ngm, pout,
                                      los_w, los_b, lom_w, lom_b,
                                      ls_w, ls_b, lm_w, lm_b, comb_w, out);
}
